// AGNOBlock_20452634263715
// MI455X (gfx1250) — compile-verified, matrix operand order checked by reading, not run on hardware
//
#include <hip/hip_runtime.h>
#include <math.h>

// ---------------------------------------------------------------------------
// MI455X (gfx1250) AGNO block: WMMA f16 edge-MLP + TDM weight staging +
// segment-softmax attention.
// ---------------------------------------------------------------------------

typedef __attribute__((ext_vector_type(16))) _Float16 v16h;
typedef __attribute__((ext_vector_type(8)))  _Float16 v8h;
typedef __attribute__((ext_vector_type(8)))  float    v8f;
typedef __attribute__((ext_vector_type(4)))  unsigned int u32x4;
typedef __attribute__((ext_vector_type(8)))  int i32x8;
typedef __attribute__((ext_vector_type(4)))  int i32x4;

#define TILE_E   64
#define THREADS  256
#define LDA      136   // padded leading dim for 128-wide f16 tiles (16B align kept)
#define LDH1     264   // padded leading dim for 256-wide f16 tiles

#if __has_builtin(__builtin_amdgcn_tensor_load_to_lds) && \
    __has_builtin(__builtin_amdgcn_s_wait_tensorcnt)
#define USE_TDM 1
#endif

// ---------------- helpers -------------------------------------------------

// Branch-free tanh-gelu: 0.5x(1+tanh(u)) == x * sigmoid(2u).
// 2u = 1.5957691216 x + 0.0713548163 x^3. Uses v_exp_f32 + v_rcp_f32 only.
__device__ __forceinline__ float gelu_fast(float x) {
  float p = x * (1.5957691216f + 0.0713548163f * x * x);
  float e = __expf(-p);
  return x * __builtin_amdgcn_rcpf(1.0f + e);
}

// Load a 16x32 f16 fragment (A-layout; B uses the mirrored layout with the
// matrix stored [N][K], i.e. "row" = output channel). Per ISA 7.12.2:
//   lanes 0-15 : row = lane,    regs 0-3 = K 0..7,  regs 4-7 = K 16..23
//   lanes 16-31: row = lane-16, regs 0-3 = K 8..15, regs 4-7 = K 24..31
__device__ __forceinline__ v16h load_frag(const _Float16* base, int ld,
                                          int row0, int k0, int lane) {
  int r  = lane & 15;
  int kb = (lane >> 4) << 3;              // 0 or 8
  const _Float16* p = base + (size_t)(row0 + r) * ld + k0 + kb;
  v8h lo = *(const v8h*)p;                // K = kb .. kb+7
  v8h hi = *(const v8h*)(p + 16);         // K = kb+16 .. kb+23
  v16h out;
#pragma unroll
  for (int i = 0; i < 8; ++i) { out[i] = lo[i]; out[i + 8] = hi[i]; }
  return out;
}

#if defined(USE_TDM)
// Tensor Data Mover: 1D copy of n8 8-byte elements global -> LDS.
// D# per ISA 8.3/8.4: group0 = {count=1, lds_addr, global_addr, type=2},
// group1 = {data_size=8B, tensor_dim0=n8, tensor_dim1=1, tile_dim0=n8,
//           tile_dim1=1, dim0_stride=n8}. Wave-level op; EXEC ignored.
// This toolchain exposes the 6-arg builtin form:
//   (u32x4 g0, i32x8 g1, i32x4 g2, i32x4 g3, i32x8 extra, i32 cpol)
__device__ __forceinline__ void tdm_load_lds(const _Float16* g, _Float16* l,
                                             unsigned n8) {
  unsigned long long ga = (unsigned long long)(size_t)g;
  unsigned lds = (unsigned)(size_t)l;     // generic->LDS offset (addr[31:0])
  u32x4 g0;
  g0[0] = 1u;                                   // count = 1 valid descriptor
  g0[1] = lds;                                  // lds_addr (bytes)
  g0[2] = (unsigned)ga;                         // global_addr[31:0]
  g0[3] = (unsigned)(ga >> 32) | (2u << 30);    // global_addr[56:32] | type=2
  i32x8 g1;
  g1[0] = (int)(3u << 16);                      // data_size=3 (8B), mask=0
  g1[1] = (int)((n8 & 0xFFFFu) << 16);          // tensor_dim0[15:0]
  g1[2] = (int)(((n8 >> 16) & 0xFFFFu) | (1u << 16)); // dim0[31:16]|dim1_lo=1
  g1[3] = (int)((n8 & 0xFFFFu) << 16);          // dim1_hi=0 | tile_dim0=n8
  g1[4] = 1;                                    // tile_dim1=1, tile_dim2=0
  g1[5] = (int)n8;                              // tensor_dim0_stride[31:0]
  g1[6] = 0;
  g1[7] = 0;
  i32x4 g2 = {0, 0, 0, 0};
  i32x4 g3 = {0, 0, 0, 0};
  i32x8 g4 = {0, 0, 0, 0, 0, 0, 0, 0};
  __builtin_amdgcn_tensor_load_to_lds(g0, g1, g2, g3, g4, 0);
}
#endif

__device__ __forceinline__ void stage_weights_fallback(const _Float16* g,
                                                       _Float16* l,
                                                       int nchunks, int tid) {
  const uint4* gs = (const uint4*)g;
  uint4* ds = (uint4*)l;
  for (int i = tid; i < nchunks; i += THREADS) ds[i] = gs[i];
}

// ---------------- small kernels -------------------------------------------

__global__ void zero_out_kernel(float* __restrict__ out, int count) {
  for (int i = blockIdx.x * blockDim.x + threadIdx.x; i < count;
       i += gridDim.x * blockDim.x)
    out[i] = 0.0f;
}

// Transpose + fp32->fp16 weight prep:
//   w1t [256][128] <- W1 [128][256]
//   w2t [128][256] <- W2 [256][128]
//   w3t [ 64][128] <- W3 [128][ 64]
__global__ void prep_weights_kernel(const float* __restrict__ W1,
                                    const float* __restrict__ W2,
                                    const float* __restrict__ W3,
                                    _Float16* __restrict__ w1t,
                                    _Float16* __restrict__ w2t,
                                    _Float16* __restrict__ w3t) {
  int i = blockIdx.x * blockDim.x + threadIdx.x;
  if (i < 256 * 128) {
    int n = i >> 7, k = i & 127;
    w1t[i] = (_Float16)W1[k * 256 + n];
  }
  if (i < 128 * 256) {
    int r = i >> 8, k = i & 255;
    w2t[i] = (_Float16)W2[k * 128 + r];
  }
  if (i < 64 * 128) {
    int r = i >> 7, k = i & 127;
    w3t[i] = (_Float16)W3[k * 64 + r];
  }
}

// One 64-thread block per node: pos-embed, Q/K projections, feature lift.
__global__ __launch_bounds__(64) void node_embed_kernel(
    const float* __restrict__ x, const float* __restrict__ y,
    const float* __restrict__ f_x,
    const float* __restrict__ peW, const float* __restrict__ peb,
    const float* __restrict__ Wq, const float* __restrict__ bq,
    const float* __restrict__ Wk, const float* __restrict__ bk,
    const float* __restrict__ liftW, const float* __restrict__ liftb,
    _Float16* __restrict__ xe16, _Float16* __restrict__ ye16,
    _Float16* __restrict__ fe16, float* __restrict__ fe32,
    float* __restrict__ Qf, float* __restrict__ Kf, int N) {
  __shared__ float sxe[32], sye[32], sfx[64];
  int n = blockIdx.x;
  if (n >= N) return;
  int c = threadIdx.x;
  if (c < 32) {
    float x0 = x[n * 3], x1 = x[n * 3 + 1], x2 = x[n * 3 + 2];
    float y0 = y[n * 3], y1 = y[n * 3 + 1], y2 = y[n * 3 + 2];
    float xv = x0 * peW[c] + x1 * peW[32 + c] + x2 * peW[64 + c] + peb[c];
    float yv = y0 * peW[c] + y1 * peW[32 + c] + y2 * peW[64 + c] + peb[c];
    sxe[c] = xv; sye[c] = yv;
    xe16[(size_t)n * 32 + c] = (_Float16)xv;
    ye16[(size_t)n * 32 + c] = (_Float16)yv;
  }
  sfx[c] = f_x[(size_t)n * 64 + c];
  __syncthreads();
  float q = bq[c], k = bk[c];
#pragma unroll 8
  for (int i = 0; i < 32; ++i) {
    q += sye[i] * Wq[i * 64 + c];
    k += sxe[i] * Wk[i * 64 + c];
  }
  Qf[(size_t)n * 64 + c] = q;
  Kf[(size_t)n * 64 + c] = k;
  float fe = liftb[c];
#pragma unroll 8
  for (int i = 0; i < 64; ++i) fe += sfx[i] * liftW[i * 64 + c];
  fe32[(size_t)n * 64 + c] = fe;
  fe16[(size_t)n * 64 + c] = (_Float16)fe;
}

__global__ void edge_scores_kernel(const int* __restrict__ src,
                                   const int* __restrict__ trg,
                                   const float* __restrict__ Qf,
                                   const float* __restrict__ Kf,
                                   float* __restrict__ sc, int E) {
  int e = blockIdx.x * blockDim.x + threadIdx.x;
  if (e >= E) return;
  const float4* q = (const float4*)(Qf + (size_t)trg[e] * 64);
  const float4* k = (const float4*)(Kf + (size_t)src[e] * 64);
  float a = 0.0f;
#pragma unroll
  for (int i = 0; i < 16; ++i) {
    float4 qv = q[i], kv = k[i];
    a += qv.x * kv.x + qv.y * kv.y + qv.z * kv.z + qv.w * kv.w;
  }
  sc[e] = a * 0.125f;  // 1/sqrt(64)
}

// trg_idx is sorted: per-node binary search gives the edge segment; compute
// segment softmax (max, sum, normalized alphas) with no float atomics.
__global__ void node_softmax_kernel(const int* __restrict__ trg,
                                    const float* __restrict__ sc,
                                    float* __restrict__ al, int N, int E) {
  int n = blockIdx.x * blockDim.x + threadIdx.x;
  if (n >= N) return;
  int lo = 0, hi = E;
  while (lo < hi) { int mid = (lo + hi) >> 1; if (trg[mid] < n) lo = mid + 1; else hi = mid; }
  int st = lo;
  hi = E;
  while (lo < hi) { int mid = (lo + hi) >> 1; if (trg[mid] < n + 1) lo = mid + 1; else hi = mid; }
  int en = lo;
  if (en <= st) return;
  float m = -3.4e38f;
  for (int e = st; e < en; ++e) m = fmaxf(m, sc[e]);
  float s = 0.0f;
  for (int e = st; e < en; ++e) s += __expf(sc[e] - m);
  float inv = 1.0f / s;
  for (int e = st; e < en; ++e) al[e] = __expf(sc[e] - m) * inv;
}

// ---------------- WMMA edge-MLP kernel ------------------------------------
// 64 edges per workgroup, 8 wave32s. Weight B-fragments are hoisted into
// registers and reused across all M-tiles; weights staged to LDS by the
// Tensor Data Mover (wave 0) with s_wait_tensorcnt + barrier handoff.

__global__ __launch_bounds__(THREADS) void edge_mlp_kernel(
    const int* __restrict__ src_idx, const int* __restrict__ trg_idx,
    const _Float16* __restrict__ xe16, const _Float16* __restrict__ ye16,
    const _Float16* __restrict__ fe16, const float* __restrict__ fe32,
    const _Float16* __restrict__ w1t, const _Float16* __restrict__ w2t,
    const _Float16* __restrict__ w3t,
    const float* __restrict__ b1, const float* __restrict__ b2,
    const float* __restrict__ b3,
    const float* __restrict__ alphas,
    float* __restrict__ out, int E) {
  __shared__ __align__(16) _Float16 sA[TILE_E * LDA];    // 17 KB
  __shared__ __align__(16) _Float16 sH1[TILE_E * LDH1];  // 33 KB
  __shared__ __align__(16) _Float16 sH2[TILE_E * LDA];   // 17 KB
  __shared__ __align__(16) _Float16 sW[256 * 128];       // 64 KB (W1t/W2t/W3t)
  __shared__ int   sSrc[TILE_E];
  __shared__ int   sTrg[TILE_E];
  __shared__ float sAl[TILE_E];

  const int tid  = threadIdx.x;
  const int lane = tid & 31;
  const int w    = tid >> 5;          // wave id 0..7
  const int base = blockIdx.x * TILE_E;

  // ---- stage W1t (TDM: 8192 x 8B) while gathering the A tile ----
#if defined(USE_TDM)
  if (w == 0) {
    tdm_load_lds(w1t, sW, 8192);
  }
#endif

  // ---- gather mlp_in tile: [ y_e[trg] | x_e[src] | f_emb[src] ] ----
  {
    int el = tid >> 2;                // edge-in-tile 0..63
    int part = tid & 3;               // 4 threads x 32 halves each
    int e = base + el;
    int ec = e < E ? e : E - 1;
    int s = src_idx[ec];
    int t = trg_idx[ec];
    if (part == 0) { sSrc[el] = s; sTrg[el] = t; sAl[el] = alphas[ec]; }
    const uint4* gsrc;
    if (part == 0)      gsrc = (const uint4*)(ye16 + (size_t)t * 32);
    else if (part == 1) gsrc = (const uint4*)(xe16 + (size_t)s * 32);
    else if (part == 2) gsrc = (const uint4*)(fe16 + (size_t)s * 64);
    else                gsrc = (const uint4*)(fe16 + (size_t)s * 64 + 32);
    uint4* ldst = (uint4*)(sA + (size_t)el * LDA + part * 32);
#pragma unroll
    for (int i = 0; i < 4; ++i) ldst[i] = gsrc[i];
  }
#if defined(USE_TDM)
  if (w == 0) __builtin_amdgcn_s_wait_tensorcnt(0);
#else
  stage_weights_fallback(w1t, sW, 4096, tid);
#endif
  __syncthreads();

  // ---- layer 1: [64x128] @ W1[128x256] + b1, gelu -> sH1 ----
  // Wave w owns output-column tiles {2w, 2w+1}; B frags held in registers.
#pragma unroll
  for (int j = 0; j < 2; ++j) {
    int nt = 2 * w + j;
    v16h bf[4];
#pragma unroll
    for (int ks = 0; ks < 4; ++ks)
      bf[ks] = load_frag(sW, 128, nt * 16, ks * 32, lane);
    int n = nt * 16 + (lane & 15);
    float bias = b1[n];
#pragma unroll
    for (int mt = 0; mt < 4; ++mt) {
      v8f acc = {};
#pragma unroll
      for (int ks = 0; ks < 4; ++ks) {
        v16h a = load_frag(sA, LDA, mt * 16, ks * 32, lane);
        acc = __builtin_amdgcn_wmma_f32_16x16x32_f16(false, a, false, bf[ks],
                                                     (short)0, acc, false, false);
      }
      int m0 = mt * 16 + ((lane >> 4) << 3);
#pragma unroll
      for (int r = 0; r < 8; ++r)
        sH1[(size_t)(m0 + r) * LDH1 + n] = (_Float16)gelu_fast(acc[r] + bias);
    }
  }
  __syncthreads();

  // ---- stage W2t ----
#if defined(USE_TDM)
  if (w == 0) {
    tdm_load_lds(w2t, sW, 8192);
    __builtin_amdgcn_s_wait_tensorcnt(0);
  }
#else
  stage_weights_fallback(w2t, sW, 4096, tid);
#endif
  __syncthreads();

  // ---- layer 2: [64x256] @ W2[256x128] + b2, gelu -> sH2 ----
  // Wave w owns output-column tile nt=w; all 8 K-step B frags held.
  {
    int nt = w;
    v16h bf[8];
#pragma unroll
    for (int ks = 0; ks < 8; ++ks)
      bf[ks] = load_frag(sW, 256, nt * 16, ks * 32, lane);
    int n = nt * 16 + (lane & 15);
    float bias = b2[n];
#pragma unroll
    for (int mt = 0; mt < 4; ++mt) {
      v8f acc = {};
#pragma unroll
      for (int ks = 0; ks < 8; ++ks) {
        v16h a = load_frag(sH1, LDH1, mt * 16, ks * 32, lane);
        acc = __builtin_amdgcn_wmma_f32_16x16x32_f16(false, a, false, bf[ks],
                                                     (short)0, acc, false, false);
      }
      int m0 = mt * 16 + ((lane >> 4) << 3);
#pragma unroll
      for (int r = 0; r < 8; ++r)
        sH2[(size_t)(m0 + r) * LDA + n] = (_Float16)gelu_fast(acc[r] + bias);
    }
  }
  __syncthreads();

  // ---- stage W3t (2048 x 8B) ----
#if defined(USE_TDM)
  if (w == 0) {
    tdm_load_lds(w3t, sW, 2048);
    __builtin_amdgcn_s_wait_tensorcnt(0);
  }
#else
  stage_weights_fallback(w3t, sW, 1024, tid);
#endif
  __syncthreads();

  // ---- layer 3: [64x128] @ W3[128x64] + b3, * f_src * alpha, scatter ----
  // 16 tiles: wave w handles nt = w&3, mt in {2*(w>>2), 2*(w>>2)+1}.
  {
    int nt  = w & 3;
    int mtb = (w >> 2) * 2;
    v16h bf[4];
#pragma unroll
    for (int ks = 0; ks < 4; ++ks)
      bf[ks] = load_frag(sW, 128, nt * 16, ks * 32, lane);
    int n = nt * 16 + (lane & 15);
    float b3n = b3[n];
#pragma unroll
    for (int mj = 0; mj < 2; ++mj) {
      int mt = mtb + mj;
      v8f acc = {};
#pragma unroll
      for (int ks = 0; ks < 4; ++ks) {
        v16h a = load_frag(sH2, LDA, mt * 16, ks * 32, lane);
        acc = __builtin_amdgcn_wmma_f32_16x16x32_f16(false, a, false, bf[ks],
                                                     (short)0, acc, false, false);
      }
      int m0 = mt * 16 + ((lane >> 4) << 3);
#pragma unroll
      for (int r = 0; r < 8; ++r) {
        int m = m0 + r;
        int e = base + m;
        if (e < E) {
          int s = sSrc[m];
          int t = sTrg[m];
          float val = sAl[m] * (acc[r] + b3n) * fe32[(size_t)s * 64 + n];
          atomicAdd(out + (size_t)t * 64 + n, val);
        }
      }
    }
  }
}

// ---------------- host launcher -------------------------------------------

extern "C" void kernel_launch(void* const* d_in, const int* in_sizes, int n_in,
                              void* d_out, int out_size, void* d_ws,
                              size_t ws_size, hipStream_t stream) {
  const float* x     = (const float*)d_in[0];
  const float* y     = (const float*)d_in[1];
  const float* f_x   = (const float*)d_in[2];
  const int*   src   = (const int*)d_in[3];
  const int*   trg   = (const int*)d_in[4];
  const float* peW   = (const float*)d_in[5];
  const float* peb   = (const float*)d_in[6];
  const float* Wq    = (const float*)d_in[7];
  const float* bq    = (const float*)d_in[8];
  const float* Wk    = (const float*)d_in[9];
  const float* bk    = (const float*)d_in[10];
  const float* liftW = (const float*)d_in[11];
  const float* liftb = (const float*)d_in[12];
  const float* W1    = (const float*)d_in[13];
  const float* b1    = (const float*)d_in[14];
  const float* W2    = (const float*)d_in[15];
  const float* b2    = (const float*)d_in[16];
  const float* W3    = (const float*)d_in[17];
  const float* b3    = (const float*)d_in[18];
  float* out = (float*)d_out;

  const int N = in_sizes[0] / 3;
  const int E = in_sizes[3];

  // Workspace carving (256B aligned chunks).
  char* ws = (char*)d_ws;
  size_t off = 0;
  auto carve = [&](size_t bytes) {
    char* p = ws + off;
    off = (off + bytes + 255) & ~(size_t)255;
    return p;
  };
  _Float16* xe16 = (_Float16*)carve((size_t)N * 32 * sizeof(_Float16));
  _Float16* ye16 = (_Float16*)carve((size_t)N * 32 * sizeof(_Float16));
  _Float16* fe16 = (_Float16*)carve((size_t)N * 64 * sizeof(_Float16));
  float*    fe32 = (float*)carve((size_t)N * 64 * sizeof(float));
  float*    Qf   = (float*)carve((size_t)N * 64 * sizeof(float));
  float*    Kf   = (float*)carve((size_t)N * 64 * sizeof(float));
  float*    sc   = (float*)carve((size_t)E * sizeof(float));
  float*    al   = (float*)carve((size_t)E * sizeof(float));
  _Float16* w1t  = (_Float16*)carve((size_t)256 * 128 * sizeof(_Float16));
  _Float16* w2t  = (_Float16*)carve((size_t)128 * 256 * sizeof(_Float16));
  _Float16* w3t  = (_Float16*)carve((size_t)64 * 128 * sizeof(_Float16));
  (void)ws_size; (void)n_in; (void)out_size;

  // 1) zero output accumulator
  zero_out_kernel<<<(N * 64 + 1023) / 1024, 1024, 0, stream>>>(out, N * 64);

  // 2) weight transpose + f16 conversion
  prep_weights_kernel<<<(256 * 128 + 255) / 256, 256, 0, stream>>>(
      W1, W2, W3, w1t, w2t, w3t);

  // 3) node embeddings / projections
  node_embed_kernel<<<N, 64, 0, stream>>>(x, y, f_x, peW, peb, Wq, bq, Wk, bk,
                                          liftW, liftb, xe16, ye16, fe16, fe32,
                                          Qf, Kf, N);

  // 4) attention scores per edge
  edge_scores_kernel<<<(E + 255) / 256, 256, 0, stream>>>(src, trg, Qf, Kf,
                                                          sc, E);

  // 5) segment softmax (sorted trg) -> alphas
  node_softmax_kernel<<<(N + 255) / 256, 256, 0, stream>>>(trg, sc, al, N, E);

  // 6) WMMA edge MLP + attention-weighted scatter
  edge_mlp_kernel<<<(E + TILE_E - 1) / TILE_E, THREADS, 0, stream>>>(
      src, trg, xe16, ye16, fe16, fe32, w1t, w2t, w3t, b1, b2, b3, al, out, E);
}
